// ConvAttention_82918638616644
// MI455X (gfx1250) — matrix operand order repaired and checked by reading, not compiled, not run on hardware
//
#include <hip/hip_runtime.h>
#include <hip/hip_bf16.h>
#include <stdint.h>

// ---------------------------------------------------------------------------
// ConvAttention (linear attention) for MI455X / gfx1250, wave32 + WMMA.
// Requires ws_size >= ~136 MB.
// ---------------------------------------------------------------------------

typedef __attribute__((ext_vector_type(16))) _Float16 v16h;
typedef __attribute__((ext_vector_type(8)))  _Float16 v8h;
typedef __attribute__((ext_vector_type(8)))  float    v8f;
typedef __attribute__((ext_vector_type(2)))  float    v2f;

#define DEVI __device__ __forceinline__

static constexpr int B_ = 8, C_ = 256, H_ = 64, W_ = 64;
static constexpr int N_ = H_ * W_;         // 4096 tokens
static constexpr int HEADS_ = 8, HID_ = 64, TMP_ = 512;

// ---- workspace layout (bytes, 256-aligned) --------------------------------
static constexpr size_t AL(size_t x) { return (x + 255) & ~(size_t)255; }
static constexpr size_t SZ_XQT  = AL((size_t)B_ * N_ * C_ * 2);     // fmap^T f16
static constexpr size_t SZ_XKVT = AL((size_t)B_ * N_ * C_ * 2);     // dwconv^T f16
static constexpr size_t SZ_WQ16 = AL((size_t)TMP_ * C_ * 2);
static constexpr size_t SZ_WKV16= AL((size_t)2 * TMP_ * C_ * 2);
static constexpr size_t SZ_WOUT16=AL((size_t)C_ * TMP_ * 2);
static constexpr size_t SZ_Q16  = AL((size_t)B_ * N_ * TMP_ * 2);   // q raw -> q softmaxed
static constexpr size_t SZ_K16  = AL((size_t)B_ * N_ * TMP_ * 2);   // k raw -> reused as att
static constexpr size_t SZ_V16  = AL((size_t)B_ * N_ * TMP_ * 2);
static constexpr size_t SZ_KMAX = AL((size_t)B_ * TMP_ * 4);
static constexpr size_t SZ_KSUM = AL((size_t)B_ * TMP_ * 4);
static constexpr size_t SZ_CTXT = AL((size_t)B_ * HEADS_ * HID_ * HID_ * 2);

static constexpr size_t OFF_XQT   = 0;
static constexpr size_t OFF_XKVT  = OFF_XQT  + SZ_XQT;
static constexpr size_t OFF_WQ16  = OFF_XKVT + SZ_XKVT;
static constexpr size_t OFF_WKV16 = OFF_WQ16 + SZ_WQ16;
static constexpr size_t OFF_WOUT16= OFF_WKV16+ SZ_WKV16;
static constexpr size_t OFF_Q16   = OFF_WOUT16 + SZ_WOUT16;
static constexpr size_t OFF_K16   = OFF_Q16  + SZ_Q16;
static constexpr size_t OFF_V16   = OFF_K16  + SZ_K16;
static constexpr size_t OFF_KMAX  = OFF_V16  + SZ_V16;
static constexpr size_t OFF_KSUM  = OFF_KMAX + SZ_KMAX;
static constexpr size_t OFF_CTXT  = OFF_KSUM + SZ_KSUM;

// ---- WMMA fragment loaders (ISA 7.12.2 layouts, wave32) -------------------
// A (16x32 f16): lane L<16 -> row M=L, K = {0..7, 16..23}; L>=16 -> K+8
DEVI v16h load_fragA(const _Float16* __restrict__ base, int ld, int row0, int k0) {
  const int lane = threadIdx.x & 31;
  const int r = row0 + (lane & 15);
  const int koff = (lane & 16) ? 8 : 0;
  const _Float16* p = base + (size_t)r * ld + k0 + koff;
  union { v16h v; v8h h[2]; } u;
  u.h[0] = *(const v8h*)(p);
  u.h[1] = *(const v8h*)(p + 16);
  return u.v;
}
// B (32x16 f16), stored as B^T[col][k] (contiguous K per col):
// lane L<16 -> col=L, K = {0..7, 16..23}; L>=16 -> K+8
DEVI v16h load_fragB(const _Float16* __restrict__ base, int ld, int col0, int k0) {
  const int lane = threadIdx.x & 31;
  const int c = col0 + (lane & 15);
  const int koff = (lane & 16) ? 8 : 0;
  const _Float16* p = base + (size_t)c * ld + k0 + koff;
  union { v16h v; v8h h[2]; } u;
  u.h[0] = *(const v8h*)(p);
  u.h[1] = *(const v8h*)(p + 16);
  return u.v;
}

DEVI v8f wmma_f16(v16h a, v16h b, v8f c) {
  return __builtin_amdgcn_wmma_f32_16x16x32_f16(false, a, false, b, (short)0, c,
                                                false, false);
}

// ---------------------------------------------------------------------------
// K0: f32 -> f16 convert (weights)
__global__ void cvt_f16_kernel(const float* __restrict__ in,
                               _Float16* __restrict__ out, int n) {
  int i = blockIdx.x * blockDim.x + threadIdx.x;
  if (i < n) out[i] = (_Float16)in[i];
}

// K1: depthwise 3x3 + transpose/convert fmap and dw to [B][N][C] f16
__global__ void __launch_bounds__(256)
prep_x_kernel(const float* __restrict__ fmap,
              const float* __restrict__ Wdw,
              _Float16* __restrict__ XqT,
              _Float16* __restrict__ XkvT) {
  const int bat = blockIdx.x >> 6;          // /64
  const int y   = blockIdx.x & 63;
  const int x   = threadIdx.x & 63;
  const int cq  = threadIdx.x >> 6;         // 0..3
  const int n   = y * 64 + x;
  for (int ii = 0; ii < 64; ++ii) {
    const int c = cq * 64 + ii;
    const float* src = fmap + ((size_t)(bat * C_ + c)) * (size_t)N_;
    const float* wc = Wdw + c * 9;
    float center = src[n];
    float s = 0.f;
#pragma unroll
    for (int dy = -1; dy <= 1; ++dy) {
#pragma unroll
      for (int dx = -1; dx <= 1; ++dx) {
        int yy = y + dy, xx = x + dx;
        if (yy >= 0 && yy < 64 && xx >= 0 && xx < 64)
          s += src[yy * 64 + xx] * wc[(dy + 1) * 3 + (dx + 1)];
      }
    }
    size_t dst = ((size_t)bat * N_ + n) * C_ + c;
    XqT[dst]  = (_Float16)center;
    XkvT[dst] = (_Float16)s;
  }
}

// K2: GEMM  out[n][o] = sum_c X[n][c] * Wt[o][c]   (WMMA f16, 64x64 tile/wave)
// oc <  512 -> outA[b][n][oc] ; oc >= 512 -> outB[b][n][oc-512]
__global__ void __launch_bounds__(256, 1)
gemm1_kernel(const _Float16* __restrict__ X,   // [B][N][256]
             const _Float16* __restrict__ Wt,  // [Mout][256]
             _Float16* __restrict__ outA,
             _Float16* __restrict__ outB, int Mout) {
  const int tilesO = Mout / 64;
  const int wavesPerBat = (N_ / 64) * tilesO;
  const int wave = blockIdx.x * (blockDim.x >> 5) + (threadIdx.x >> 5);
  if (wave >= B_ * wavesPerBat) return;
  const int bat = wave / wavesPerBat;
  const int rem = wave % wavesPerBat;
  const int o0 = (rem % tilesO) * 64;
  const int n0 = (rem / tilesO) * 64;
  const _Float16* Xb = X + (size_t)bat * N_ * C_;

  v8f acc[4][4] = {};
#pragma unroll 1
  for (int k0 = 0; k0 < C_; k0 += 32) {
    v16h a0 = load_fragA(Xb, C_, n0 +  0, k0);
    v16h a1 = load_fragA(Xb, C_, n0 + 16, k0);
    v16h a2 = load_fragA(Xb, C_, n0 + 32, k0);
    v16h a3 = load_fragA(Xb, C_, n0 + 48, k0);
    if (k0 + 32 < C_) {
      __builtin_prefetch(Xb + (size_t)(n0 + (threadIdx.x & 31)) * C_ + k0 + 32, 0, 1);
    }
#pragma unroll
    for (int j = 0; j < 4; ++j) {
      v16h bb = load_fragB(Wt, C_, o0 + 16 * j, k0);
      acc[0][j] = wmma_f16(a0, bb, acc[0][j]);
      acc[1][j] = wmma_f16(a1, bb, acc[1][j]);
      acc[2][j] = wmma_f16(a2, bb, acc[2][j]);
      acc[3][j] = wmma_f16(a3, bb, acc[3][j]);
    }
  }
  const int lane = threadIdx.x & 31;
  const int nn = lane & 15;
  const int hi = (lane & 16) ? 8 : 0;
#pragma unroll
  for (int j = 0; j < 4; ++j) {
    const int oc = o0 + 16 * j + nn;
#pragma unroll
    for (int i = 0; i < 4; ++i)
#pragma unroll
      for (int r = 0; r < 8; ++r) {
        const int tok = n0 + 16 * i + r + hi;
        const float val = acc[i][j][r];
        if (oc < TMP_)
          outA[((size_t)bat * N_ + tok) * TMP_ + oc] = (_Float16)val;
        else
          outB[((size_t)bat * N_ + tok) * TMP_ + (oc - TMP_)] = (_Float16)val;
      }
  }
}

// K3: per-(b,o) max and 1/sum(exp) over 4096 tokens (k softmax over n)
__global__ void __launch_bounds__(256)
kstats_kernel(const _Float16* __restrict__ k16,
              float* __restrict__ kmax,
              float* __restrict__ ksum) {
  const int bat = blockIdx.x >> 9;   // /512
  const int o   = blockIdx.x & 511;
  const _Float16* col = k16 + (size_t)bat * N_ * TMP_ + o;
  __shared__ float red[256];
  float m = -1e30f;
  for (int n = threadIdx.x; n < N_; n += 256)
    m = fmaxf(m, (float)col[(size_t)n * TMP_]);
  red[threadIdx.x] = m; __syncthreads();
  for (int s = 128; s; s >>= 1) {
    if ((int)threadIdx.x < s) red[threadIdx.x] = fmaxf(red[threadIdx.x], red[threadIdx.x + s]);
    __syncthreads();
  }
  m = red[0]; __syncthreads();
  float sum = 0.f;
  for (int n = threadIdx.x; n < N_; n += 256)
    sum += __expf((float)col[(size_t)n * TMP_] - m);
  red[threadIdx.x] = sum; __syncthreads();
  for (int s = 128; s; s >>= 1) {
    if ((int)threadIdx.x < s) red[threadIdx.x] += red[threadIdx.x + s];
    __syncthreads();
  }
  if (threadIdx.x == 0) {
    kmax[bat * TMP_ + o] = m;
    ksum[bat * TMP_ + o] = 1.0f / red[0];
  }
}

// K4: ctx^T[e][d] = sum_n v[n,e] * softmax_n(k)[n,d]   (WMMA f32 16x16x4)
// one block (8 waves) per (b,h); each wave owns a 512-token K-chunk.
__global__ void __launch_bounds__(256, 1)
ctx_kernel(const _Float16* __restrict__ k16,
           const _Float16* __restrict__ v16,
           const float* __restrict__ kmax,
           const float* __restrict__ ksum,
           _Float16* __restrict__ ctxT) {
  const int bat = blockIdx.x >> 3;
  const int h   = blockIdx.x & 7;
  const int wave = threadIdx.x >> 5;
  const int lane = threadIdx.x & 31;
  __shared__ float sctx[HID_ * HID_];
  for (int i = threadIdx.x; i < HID_ * HID_; i += 256) sctx[i] = 0.f;
  __syncthreads();

  const int koff = (lane & 16) ? 2 : 0;
  const int l15 = lane & 15;
  float cm[4], cs[4];
#pragma unroll
  for (int j = 0; j < 4; ++j) {
    const int bo = bat * TMP_ + h * HID_ + 16 * j + l15;
    cm[j] = kmax[bo]; cs[j] = ksum[bo];
  }
  const size_t baseBH = (size_t)bat * N_ * TMP_ + h * HID_;

  v8f acc[4][4] = {};
#pragma unroll 1
  for (int kk = 0; kk < 512; kk += 4) {
    const int n0 = wave * 512 + kk + koff;
    const size_t r0 = baseBH + (size_t)n0 * TMP_;
    const size_t r1 = r0 + TMP_;
    v2f a[4];
#pragma unroll
    for (int i = 0; i < 4; ++i) {            // A rows = e (from v)
      const int e = 16 * i + l15;
      a[i].x = (float)v16[r0 + e];
      a[i].y = (float)v16[r1 + e];
    }
#pragma unroll
    for (int j = 0; j < 4; ++j) {            // B cols = d (softmaxed k)
      const int d = 16 * j + l15;
      v2f bf;
      bf.x = __expf((float)k16[r0 + d] - cm[j]) * cs[j];
      bf.y = __expf((float)k16[r1 + d] - cm[j]) * cs[j];
#pragma unroll
      for (int i = 0; i < 4; ++i)
        acc[i][j] = __builtin_amdgcn_wmma_f32_16x16x4_f32(
            false, a[i], false, bf, (short)0, acc[i][j], false, false);
    }
  }
  const int hi = (lane & 16) ? 8 : 0;
#pragma unroll
  for (int i = 0; i < 4; ++i)
#pragma unroll
    for (int j = 0; j < 4; ++j)
#pragma unroll
      for (int r = 0; r < 8; ++r) {
        const int e = 16 * i + r + hi, d = 16 * j + l15;
        atomicAdd(&sctx[e * HID_ + d], acc[i][j][r]);
      }
  __syncthreads();
  _Float16* dst = ctxT + (size_t)(bat * HEADS_ + h) * HID_ * HID_;
  for (int i = threadIdx.x; i < HID_ * HID_; i += 256)
    dst[i] = (_Float16)sctx[i];
}

// K5: in-place softmax over d (per b,n,h) * scale      (one wave per head)
__global__ void __launch_bounds__(256)
qsoftmax_kernel(_Float16* __restrict__ q16) {
  const int bat = blockIdx.x >> 12;
  const int n   = blockIdx.x & 4095;
  const int h   = threadIdx.x >> 5;
  const int lane = threadIdx.x & 31;
  _Float16* p = q16 + ((size_t)bat * N_ + n) * TMP_ + h * HID_;
  float x0 = (float)p[lane], x1 = (float)p[lane + 32];
  float m = fmaxf(x0, x1);
  for (int off = 16; off; off >>= 1) m = fmaxf(m, __shfl_xor(m, off, 32));
  float e0 = __expf(x0 - m), e1 = __expf(x1 - m);
  float s = e0 + e1;
  for (int off = 16; off; off >>= 1) s += __shfl_xor(s, off, 32);
  const float r = 0.125f / s;   // scale = hid^-0.5 = 1/8
  p[lane]      = (_Float16)(e0 * r);
  p[lane + 32] = (_Float16)(e1 * r);
}

// K6: att[n][h*64+e] = GELU( sum_d qs[n,d] * ctxT[e][d] )   (WMMA f16, K=64)
__global__ void __launch_bounds__(256, 1)
att_kernel(const _Float16* __restrict__ q16,
           const _Float16* __restrict__ ctxT,
           _Float16* __restrict__ att) {
  const int wave = blockIdx.x * (blockDim.x >> 5) + (threadIdx.x >> 5);
  if (wave >= B_ * HEADS_ * (N_ / 64)) return;
  const int bh = wave / (N_ / 64);
  const int n0 = (wave % (N_ / 64)) * 64;
  const int bat = bh >> 3, h = bh & 7;
  const _Float16* qb = q16 + (size_t)bat * N_ * TMP_ + h * HID_;
  const _Float16* cb = ctxT + (size_t)bh * HID_ * HID_;

  v8f acc[4][4] = {};
#pragma unroll 1
  for (int k0 = 0; k0 < HID_; k0 += 32) {
    v16h a0 = load_fragA(qb, TMP_, n0 +  0, k0);
    v16h a1 = load_fragA(qb, TMP_, n0 + 16, k0);
    v16h a2 = load_fragA(qb, TMP_, n0 + 32, k0);
    v16h a3 = load_fragA(qb, TMP_, n0 + 48, k0);
#pragma unroll
    for (int j = 0; j < 4; ++j) {
      v16h bb = load_fragB(cb, HID_, 16 * j, k0);
      acc[0][j] = wmma_f16(a0, bb, acc[0][j]);
      acc[1][j] = wmma_f16(a1, bb, acc[1][j]);
      acc[2][j] = wmma_f16(a2, bb, acc[2][j]);
      acc[3][j] = wmma_f16(a3, bb, acc[3][j]);
    }
  }
  const int lane = threadIdx.x & 31;
  const int l15 = lane & 15, hi = (lane & 16) ? 8 : 0;
#pragma unroll
  for (int j = 0; j < 4; ++j) {
    const int e = 16 * j + l15;
#pragma unroll
    for (int i = 0; i < 4; ++i)
#pragma unroll
      for (int r = 0; r < 8; ++r) {
        const int tok = n0 + 16 * i + r + hi;
        const float x = acc[i][j][r];
        const float g = 0.5f * x * (1.0f + erff(x * 0.70710678118f));
        att[((size_t)bat * N_ + tok) * TMP_ + h * HID_ + e] = (_Float16)g;
      }
  }
}

// K7: out[b][co][n] = sum_o att[n][o] * Wout[co][o] + bout[co]  (WMMA, K=512)
__global__ void __launch_bounds__(256, 1)
outgemm_kernel(const _Float16* __restrict__ att,
               const _Float16* __restrict__ Wout16,
               const float* __restrict__ bout,
               float* __restrict__ out) {
  const int tilesO = C_ / 64;                       // 4
  const int wavesPerBat = (N_ / 64) * tilesO;       // 256
  const int wave = blockIdx.x * (blockDim.x >> 5) + (threadIdx.x >> 5);
  if (wave >= B_ * wavesPerBat) return;
  const int bat = wave / wavesPerBat;
  const int rem = wave % wavesPerBat;
  const int co0 = (rem % tilesO) * 64;
  const int n0 = (rem / tilesO) * 64;
  const _Float16* ab = att + (size_t)bat * N_ * TMP_;

  v8f acc[4][4] = {};
#pragma unroll 1
  for (int k0 = 0; k0 < TMP_; k0 += 32) {
    v16h a0 = load_fragA(ab, TMP_, n0 +  0, k0);
    v16h a1 = load_fragA(ab, TMP_, n0 + 16, k0);
    v16h a2 = load_fragA(ab, TMP_, n0 + 32, k0);
    v16h a3 = load_fragA(ab, TMP_, n0 + 48, k0);
    if (k0 + 32 < TMP_) {
      __builtin_prefetch(ab + (size_t)(n0 + (threadIdx.x & 31)) * TMP_ + k0 + 32, 0, 1);
    }
#pragma unroll
    for (int j = 0; j < 4; ++j) {
      v16h bb = load_fragB(Wout16, TMP_, co0 + 16 * j, k0);
      acc[0][j] = wmma_f16(a0, bb, acc[0][j]);
      acc[1][j] = wmma_f16(a1, bb, acc[1][j]);
      acc[2][j] = wmma_f16(a2, bb, acc[2][j]);
      acc[3][j] = wmma_f16(a3, bb, acc[3][j]);
    }
  }
  const int lane = threadIdx.x & 31;
  const int l15 = lane & 15, hi = (lane & 16) ? 8 : 0;
#pragma unroll
  for (int j = 0; j < 4; ++j) {
    const int co = co0 + 16 * j + l15;
    const float bias = bout[co];
    float* ob = out + ((size_t)bat * C_ + co) * N_;
#pragma unroll
    for (int i = 0; i < 4; ++i)
#pragma unroll
      for (int r = 0; r < 8; ++r)
        ob[n0 + 16 * i + r + hi] = acc[i][j][r] + bias;
  }
}

// ---------------------------------------------------------------------------
extern "C" void kernel_launch(void* const* d_in, const int* in_sizes, int n_in,
                              void* d_out, int out_size, void* d_ws, size_t ws_size,
                              hipStream_t stream) {
  (void)in_sizes; (void)n_in; (void)out_size; (void)ws_size;
  const float* fmap = (const float*)d_in[0];
  const float* Wq   = (const float*)d_in[1];
  const float* Wdw  = (const float*)d_in[2];
  const float* Wkv  = (const float*)d_in[3];
  const float* Wout = (const float*)d_in[4];
  const float* bout = (const float*)d_in[5];
  float* out = (float*)d_out;

  uint8_t* w = (uint8_t*)d_ws;
  _Float16* XqT    = (_Float16*)(w + OFF_XQT);
  _Float16* XkvT   = (_Float16*)(w + OFF_XKVT);
  _Float16* Wq16   = (_Float16*)(w + OFF_WQ16);
  _Float16* Wkv16  = (_Float16*)(w + OFF_WKV16);
  _Float16* Wout16 = (_Float16*)(w + OFF_WOUT16);
  _Float16* q16    = (_Float16*)(w + OFF_Q16);
  _Float16* k16    = (_Float16*)(w + OFF_K16);   // later reused for att
  _Float16* v16    = (_Float16*)(w + OFF_V16);
  float*    kmax   = (float*)(w + OFF_KMAX);
  float*    ksum   = (float*)(w + OFF_KSUM);
  _Float16* ctxT   = (_Float16*)(w + OFF_CTXT);

  // weights -> f16
  cvt_f16_kernel<<<(TMP_ * C_ + 255) / 256, 256, 0, stream>>>(Wq, Wq16, TMP_ * C_);
  cvt_f16_kernel<<<(2 * TMP_ * C_ + 255) / 256, 256, 0, stream>>>(Wkv, Wkv16, 2 * TMP_ * C_);
  cvt_f16_kernel<<<(C_ * TMP_ + 255) / 256, 256, 0, stream>>>(Wout, Wout16, C_ * TMP_);

  // depthwise conv + token-major transposes
  prep_x_kernel<<<B_ * H_, 256, 0, stream>>>(fmap, Wdw, XqT, XkvT);

  // q = Wq * fmap ; [k|v] = Wkv * dw
  gemm1_kernel<<<(B_ * 64 * 8) / 8, 256, 0, stream>>>(XqT, Wq16, q16, nullptr, TMP_);
  gemm1_kernel<<<(B_ * 64 * 16) / 8, 256, 0, stream>>>(XkvT, Wkv16, k16, v16, 2 * TMP_);

  // softmax stats over token dim for k
  kstats_kernel<<<B_ * TMP_, 256, 0, stream>>>(k16, kmax, ksum);

  // ctx^T per (b,h)
  ctx_kernel<<<B_ * HEADS_, 256, 0, stream>>>(k16, v16, kmax, ksum, ctxT);

  // softmax over feature dim for q (in place, * scale)
  qsoftmax_kernel<<<B_ * N_, 256, 0, stream>>>(q16);

  // att = GELU(q @ ctx)  (writes into k16 buffer, k no longer needed)
  att_kernel<<<(B_ * HEADS_ * 64) / 8, 256, 0, stream>>>(q16, ctxT, k16);

  // final projection + bias -> NCHW f32
  outgemm_kernel<<<(B_ * 256) / 8, 256, 0, stream>>>(k16, Wout16, bout, out);
}
